// StructureAE_51685636440165
// MI455X (gfx1250) — matrix operand order repaired and checked by reading, not compiled, not run on hardware
//
#include <hip/hip_runtime.h>
#include <hip/hip_bf16.h>
#include <cstdint>

#define N_NODES   10000
#define IN_DIM    512
#define OUT_DIM   128
#define SLOPE_ATT 0.5f
#define SLOPE_ACT 0.01f

typedef __attribute__((ext_vector_type(8)))  float  v8f;
typedef __attribute__((ext_vector_type(16))) __bf16 v16bf;
typedef __attribute__((ext_vector_type(8)))  __bf16 v8bf;

__device__ __forceinline__ unsigned short f2bf(float f) {  // RNE fp32 -> bf16
  unsigned u = __float_as_uint(f);
  u += 0x7FFFu + ((u >> 16) & 1u);
  return (unsigned short)(u >> 16);
}
__device__ __forceinline__ float sigmoidf_(float v) {
  return 1.0f / (1.0f + __expf(-v));
}

// ---------------------------------------------------------------------------
// K1: h = x @ W (one block per node, 128 threads = one out channel each),
//     plus a_src = h.att_src, a_dst = h.att_dst via LDS reduction.
// ---------------------------------------------------------------------------
__global__ __launch_bounds__(128)
void k_gemm_h(const float* __restrict__ x, const float* __restrict__ W,
              const float* __restrict__ att_src, const float* __restrict__ att_dst,
              float* __restrict__ h, float* __restrict__ a_src, float* __restrict__ a_dst) {
  __shared__ float xs[IN_DIM];
  __shared__ float red[OUT_DIM];
  const int node = blockIdx.x;
  const int c    = threadIdx.x;            // 0..127
  for (int k = c; k < IN_DIM; k += OUT_DIM) xs[k] = x[(size_t)node * IN_DIM + k];
  __syncthreads();
  float acc = 0.0f;
#pragma unroll 8
  for (int k = 0; k < IN_DIM; ++k) acc = fmaf(xs[k], W[(size_t)k * OUT_DIM + c], acc);
  h[(size_t)node * OUT_DIM + c] = acc;

  red[c] = acc * att_src[c];
  __syncthreads();
  for (int s = OUT_DIM / 2; s > 0; s >>= 1) {
    if (c < s) red[c] += red[c + s];
    __syncthreads();
  }
  if (c == 0) a_src[node] = red[0];
  __syncthreads();
  red[c] = acc * att_dst[c];
  __syncthreads();
  for (int s = OUT_DIM / 2; s > 0; s >>= 1) {
    if (c < s) red[c] += red[c + s];
    __syncthreads();
  }
  if (c == 0) a_dst[node] = red[0];
}

// ---------------------------------------------------------------------------
// K2a: incoming-degree histogram over messages (edges + self loops)
// ---------------------------------------------------------------------------
__global__ __launch_bounds__(256)
void k_deg(const int* __restrict__ ei, int E, int M, int* __restrict__ deg) {
  int m = blockIdx.x * 256 + threadIdx.x;
  if (m >= M) return;
  int d = (m < E) ? ei[E + m] : (m - E);
  atomicAdd(&deg[d], 1);
}

// ---------------------------------------------------------------------------
// K2b: exclusive scan of deg -> rowptr (single 256-thread block)
// ---------------------------------------------------------------------------
__global__ __launch_bounds__(256)
void k_scan(const int* __restrict__ deg, int* __restrict__ rowptr) {
  __shared__ int part[256];
  const int t  = threadIdx.x;
  const int CH = 40;                         // 256*40 = 10240 >= N_NODES
  const int base = t * CH;
  int sum = 0;
  for (int i = 0; i < CH; ++i) {
    int idx = base + i;
    if (idx < N_NODES) sum += deg[idx];
  }
  part[t] = sum;
  __syncthreads();
  for (int off = 1; off < 256; off <<= 1) {  // inclusive Hillis-Steele
    int v = (t >= off) ? part[t - off] : 0;
    __syncthreads();
    part[t] += v;
    __syncthreads();
  }
  int prefix = (t == 0) ? 0 : part[t - 1];
  for (int i = 0; i < CH; ++i) {
    int idx = base + i;
    if (idx < N_NODES) { rowptr[idx] = prefix; prefix += deg[idx]; }
  }
  if (t == 255) rowptr[N_NODES] = prefix;    // = M
}

// ---------------------------------------------------------------------------
// K2c: scatter message source ids into CSR order (per-dst cursors)
// ---------------------------------------------------------------------------
__global__ __launch_bounds__(256)
void k_csr(const int* __restrict__ ei, int E, int M,
           const int* __restrict__ rowptr, int* __restrict__ cursor,
           int* __restrict__ csr_src) {
  int m = blockIdx.x * 256 + threadIdx.x;
  if (m >= M) return;
  int s, d;
  if (m < E) { s = ei[m]; d = ei[E + m]; } else { s = d = m - E; }
  int pos = rowptr[d] + atomicAdd(&cursor[d], 1);
  csr_src[pos] = s;
}

// ---------------------------------------------------------------------------
// K3: per-dst gather: segment max -> softmax denom -> Sum alpha*h[src],
//     then bias + leaky(0.01) + bf16 convert. One 128-thread block per node.
//     No float atomics; h reads fully coalesced over channels.
// ---------------------------------------------------------------------------
#define GCHUNK 128
__global__ __launch_bounds__(128)
void k_gather(const int* __restrict__ rowptr, const int* __restrict__ csr_src,
              const float* __restrict__ a_src, const float* __restrict__ a_dst,
              const float* __restrict__ h, const float* __restrict__ bias,
              float* __restrict__ embed_out, unsigned short* __restrict__ ebf) {
  __shared__ float red[128];
  __shared__ float s_alpha[GCHUNK];
  __shared__ int   s_src[GCHUNK];
  const int dst = blockIdx.x;
  const int t   = threadIdx.x;               // also = output channel
  const int beg = rowptr[dst];
  const int end = rowptr[dst + 1];
  const float ad = a_dst[dst];

  // pass 1: segment max of leaky(a_src[s]+a_dst[d], 0.5)
  float lmax = -3.4e38f;
  for (int i = beg + t; i < end; i += 128) {
    float v  = a_src[csr_src[i]] + ad;
    float lv = (v >= 0.0f) ? v : SLOPE_ATT * v;
    lmax = fmaxf(lmax, lv);
  }
  red[t] = lmax;
  __syncthreads();
  for (int s = 64; s > 0; s >>= 1) {
    if (t < s) red[t] = fmaxf(red[t], red[t + s]);
    __syncthreads();
  }
  const float m = red[0];
  __syncthreads();

  // pass 2: softmax denominator
  float lsum = 0.0f;
  for (int i = beg + t; i < end; i += 128) {
    float v  = a_src[csr_src[i]] + ad;
    float lv = (v >= 0.0f) ? v : SLOPE_ATT * v;
    lsum += __expf(lv - m);
  }
  red[t] = lsum;
  __syncthreads();
  for (int s = 64; s > 0; s >>= 1) {
    if (t < s) red[t] += red[t + s];
    __syncthreads();
  }
  const float inv_denom = 1.0f / red[0];
  __syncthreads();

  // pass 3: chunked accumulate; (src, alpha) staged in LDS per chunk
  float acc = 0.0f;
  for (int cbeg = beg; cbeg < end; cbeg += GCHUNK) {
    int i = cbeg + t;
    if (i < end) {
      int s = csr_src[i];
      float v  = a_src[s] + ad;
      float lv = (v >= 0.0f) ? v : SLOPE_ATT * v;
      s_alpha[t] = __expf(lv - m) * inv_denom;
      s_src[t]   = s;
    }
    __syncthreads();
    int cnt = min(GCHUNK, end - cbeg);
    for (int k = 0; k < cnt; ++k)
      acc = fmaf(s_alpha[k], h[(size_t)s_src[k] * OUT_DIM + t], acc);
    __syncthreads();
  }

  float v = acc + bias[t];
  v = (v >= 0.0f) ? v : SLOPE_ACT * v;
  unsigned idx = (unsigned)dst * OUT_DIM + t;
  embed_out[idx] = v;
  ebf[idx] = f2bf(v);
}

// ---------------------------------------------------------------------------
// K4: recon = sigmoid(embed @ embed^T) via v_wmma_f32_16x16x32_bf16.
// Block = 8 waves, tile 128(i) x 64(j). B tile (64 rows x full K=128) is
// staged once per block in LDS (row stride 136 elems = 272B -> 16-lane
// ds_load_b128 groups stripe across all 64 banks). Wave w owns rows
// [i0+w*16,+16) and all 64 j columns (4 accumulators).
// A-lane order used here (same as B): lane m<16: K{kt..kt+15},
//   lanes 16-31: K{kt+16..kt+31}. B-lane: lane n holds row E[j0+n], 16 contiguous
//   K at kt (lanes<16) / kt+16 (lanes>=16). C VGPR r: M=r (+8 upper), N=lane&15.
// Indices clamped (EXEC stays all-1s for WMMA); stores guarded.
// ---------------------------------------------------------------------------
#define TJ 64
#define LDSTRIDE 136   // 128 + 8 pad bf16 elems per row (272 B)
__global__ __launch_bounds__(256)
void k_recon(const unsigned short* __restrict__ Ebf, float* __restrict__ recon) {
  __shared__ unsigned short Bs[TJ * LDSTRIDE];   // 17408 B
  const int t    = threadIdx.x;
  const int lane = t & 31;
  const int wave = t >> 5;
  const int i0   = blockIdx.y * 128 + wave * 16;
  const int j0   = blockIdx.x * TJ;
  const int half = lane >> 4;     // 0 = lanes 0-15, 1 = lanes 16-31
  const int l16  = lane & 15;

  // cooperative stage of B tile: 64 rows x 128 elems, 16B chunks (4/thread)
  for (int q = t; q < TJ * 16; q += 256) {
    int row  = q >> 4;
    int cc   = q & 15;
    int grow = j0 + row; if (grow >= N_NODES) grow = N_NODES - 1;
    *reinterpret_cast<v8bf*>(&Bs[row * LDSTRIDE + cc * 8]) =
        *reinterpret_cast<const v8bf*>(Ebf + (size_t)grow * OUT_DIM + cc * 8);
  }
  __syncthreads();

  int arow = i0 + l16; if (arow >= N_NODES) arow = N_NODES - 1;
  const unsigned short* Arow = Ebf + (size_t)arow * OUT_DIM;

  v8f c0 = {}, c1 = {}, c2 = {}, c3 = {};

#pragma unroll
  for (int kt = 0; kt < OUT_DIM; kt += 32) {
    const int ka = kt + half * 16;       // A: 16 contiguous elems at ka
    union { v16bf v; v8bf h[2]; } a;
    a.h[0] = *reinterpret_cast<const v8bf*>(Arow + ka);
    a.h[1] = *reinterpret_cast<const v8bf*>(Arow + ka + 8);

    const int kb = kt + half * 16;       // B: 16 contiguous elems at kb
    union { v16bf v; v8bf h[2]; } b0, b1, b2, b3;
    b0.h[0] = *reinterpret_cast<const v8bf*>(&Bs[( 0 + l16) * LDSTRIDE + kb]);
    b0.h[1] = *reinterpret_cast<const v8bf*>(&Bs[( 0 + l16) * LDSTRIDE + kb + 8]);
    b1.h[0] = *reinterpret_cast<const v8bf*>(&Bs[(16 + l16) * LDSTRIDE + kb]);
    b1.h[1] = *reinterpret_cast<const v8bf*>(&Bs[(16 + l16) * LDSTRIDE + kb + 8]);
    b2.h[0] = *reinterpret_cast<const v8bf*>(&Bs[(32 + l16) * LDSTRIDE + kb]);
    b2.h[1] = *reinterpret_cast<const v8bf*>(&Bs[(32 + l16) * LDSTRIDE + kb + 8]);
    b3.h[0] = *reinterpret_cast<const v8bf*>(&Bs[(48 + l16) * LDSTRIDE + kb]);
    b3.h[1] = *reinterpret_cast<const v8bf*>(&Bs[(48 + l16) * LDSTRIDE + kb + 8]);

    c0 = __builtin_amdgcn_wmma_f32_16x16x32_bf16(false, a.v, false, b0.v, (short)0, c0, false, false);
    c1 = __builtin_amdgcn_wmma_f32_16x16x32_bf16(false, a.v, false, b1.v, (short)0, c1, false, false);
    c2 = __builtin_amdgcn_wmma_f32_16x16x32_bf16(false, a.v, false, b2.v, (short)0, c2, false, false);
    c3 = __builtin_amdgcn_wmma_f32_16x16x32_bf16(false, a.v, false, b3.v, (short)0, c3, false, false);
  }

#pragma unroll
  for (int r = 0; r < 8; ++r) {
    int orow = i0 + half * 8 + r;        // C layout: VGPR r -> M=r (+8 upper)
    if (orow < N_NODES) {
      float* outrow = recon + (size_t)orow * N_NODES;
      int jc = j0 + l16;
      if (jc      < N_NODES) outrow[jc]      = sigmoidf_(c0[r]);
      if (jc + 16 < N_NODES) outrow[jc + 16] = sigmoidf_(c1[r]);
      if (jc + 32 < N_NODES) outrow[jc + 32] = sigmoidf_(c2[r]);
      if (jc + 48 < N_NODES) outrow[jc + 48] = sigmoidf_(c3[r]);
    }
  }
}

// ---------------------------------------------------------------------------
extern "C" void kernel_launch(void* const* d_in, const int* in_sizes, int n_in,
                              void* d_out, int out_size, void* d_ws, size_t ws_size,
                              hipStream_t stream) {
  const float* x       = (const float*)d_in[0];
  const int*   ei      = (const int*)  d_in[1];   // [2,E] row-major int32
  const float* W       = (const float*)d_in[2];
  const float* att_src = (const float*)d_in[3];
  const float* att_dst = (const float*)d_in[4];
  const float* bias    = (const float*)d_in[5];

  const int E = in_sizes[1] / 2;
  const int M = E + N_NODES;                      // edges + self loops

  // workspace carve-out (256B aligned)
  char* p = (char*)d_ws;
  auto carve = [&](size_t bytes) -> void* {
    void* r = (void*)p;
    p += (bytes + 255) & ~(size_t)255;
    return r;
  };
  float*          h       = (float*)         carve((size_t)N_NODES * OUT_DIM * 4);
  float*          a_src   = (float*)         carve((size_t)N_NODES * 4);
  float*          a_dst   = (float*)         carve((size_t)N_NODES * 4);
  int*            deg     = (int*)           carve((size_t)N_NODES * 4);
  int*            rowptr  = (int*)           carve((size_t)(N_NODES + 1) * 4);
  int*            cursor  = (int*)           carve((size_t)N_NODES * 4);
  int*            csr_src = (int*)           carve((size_t)M * 4);
  unsigned short* ebf     = (unsigned short*)carve((size_t)N_NODES * OUT_DIM * 2);

  float* recon     = (float*)d_out;                         // [N,N]
  float* embed_out = recon + (size_t)N_NODES * N_NODES;     // [N,128]

  hipMemsetAsync(deg,    0, (size_t)N_NODES * 4, stream);
  hipMemsetAsync(cursor, 0, (size_t)N_NODES * 4, stream);

  k_gemm_h<<<N_NODES, 128, 0, stream>>>(x, W, att_src, att_dst, h, a_src, a_dst);

  const int mb = (M + 255) / 256;
  k_deg <<<mb, 256, 0, stream>>>(ei, E, M, deg);
  k_scan<<<1, 256, 0, stream>>>(deg, rowptr);
  k_csr <<<mb, 256, 0, stream>>>(ei, E, M, rowptr, cursor, csr_src);

  k_gather<<<N_NODES, 128, 0, stream>>>(rowptr, csr_src, a_src, a_dst,
                                        h, bias, embed_out, ebf);

  dim3 grid((N_NODES + TJ - 1) / TJ, (N_NODES + 127) / 128);  // (157, 79)
  k_recon<<<grid, 256, 0, stream>>>(ebf, recon);
}
